// MoELayer_8555574854061
// MI455X (gfx1250) — compile-verified, matrix operand order checked by reading, not run on hardware
//
#include <hip/hip_runtime.h>
#include <hip/hip_bf16.h>

typedef __attribute__((ext_vector_type(16))) _Float16 v16h;
typedef __attribute__((ext_vector_type(8)))  float    v8f;

#define D_DIM 1024
#define H_DIM 1024
#define E_NUM 8
#define BM 128
#define BN 128
#define BKS 64          // K halves staged per pipeline stage (2 WMMA substeps)
#define LDK2 72         // padded K stride in halves (64 + 8) -> 144B rows, 16B aligned
#define NSTAGE (D_DIM / BKS)

#define AS3 __attribute__((address_space(3)))

// LDS byte offset (what GLOBAL_LOAD_ASYNC_TO_LDS wants in its VDST VGPR)
static __device__ __forceinline__ unsigned lds_off(const void* p) {
  return (unsigned)(unsigned long long)(AS3 const void*)p;
}

// async 16B global -> LDS copy, tracked by ASYNCcnt (CDNA5)
static __device__ __forceinline__ void async_copy_b128(unsigned lds_byte_off,
                                                       const unsigned short* gsrc) {
  asm volatile("global_load_async_to_lds_b128 %0, %1, off"
               :: "v"(lds_byte_off), "v"(gsrc) : "memory");
}

// ---------------------------------------------------------------- zero init
__global__ void zero_kernel(float* __restrict__ out, size_t n,
                            int* __restrict__ cnt, float* __restrict__ ent) {
  size_t i = (size_t)blockIdx.x * blockDim.x + threadIdx.x;
  if (i < n) out[i] = 0.0f;
  if (i < E_NUM) cnt[i] = 0;
  if (i == 0) *ent = 0.0f;
}

// ------------------------------------------- expert_w f32[E][D][H] -> f16[E][H][D]
__global__ void wcvt_kernel(const float* __restrict__ w, unsigned short* __restrict__ wt) {
  __shared__ _Float16 tile[32][33];
  int e  = blockIdx.z;
  int h0 = blockIdx.x * 32;
  int d0 = blockIdx.y * 32;
  int tx = threadIdx.x & 31;
  int ty = threadIdx.x >> 5;   // 8 row-groups
  for (int r = ty; r < 32; r += 8)
    tile[r][tx] = (_Float16)w[((size_t)e * D_DIM + d0 + r) * H_DIM + h0 + tx];
  __syncthreads();
  for (int r = ty; r < 32; r += 8) {
    union { _Float16 h; unsigned short u; } cv;
    cv.h = tile[tx][r];
    wt[((size_t)e * H_DIM + h0 + r) * D_DIM + d0 + tx] = cv.u;
  }
}

// ------------------------- gate: logits, softmax, entropy, top-2, token lists, x->f16
__global__ void gate_kernel(const float* __restrict__ x,
                            const float* __restrict__ gw,
                            const float* __restrict__ gb,
                            unsigned short* __restrict__ xh,
                            int* __restrict__ tok_list,
                            float* __restrict__ prob_list,
                            int* __restrict__ cnt,
                            float* __restrict__ ent_accum,
                            int N) {
  int tok  = (int)((blockIdx.x * (size_t)blockDim.x + threadIdx.x) >> 5);
  int lane = threadIdx.x & 31;
  if (tok >= N) return;
  const float* xr = x + (size_t)tok * D_DIM;
  unsigned short* xhr = xh + (size_t)tok * D_DIM;

  float acc[E_NUM];
#pragma unroll
  for (int e = 0; e < E_NUM; ++e) acc[e] = 0.0f;

  for (int i = 0; i < D_DIM / 32; ++i) {
    int d = lane + 32 * i;
    float v = xr[d];
    union { _Float16 h; unsigned short u; } cv;
    cv.h = (_Float16)v;
    xhr[d] = cv.u;
    const float4* g = (const float4*)(gw + (size_t)d * E_NUM);
    float4 g0 = g[0], g1 = g[1];
    acc[0] += v * g0.x; acc[1] += v * g0.y; acc[2] += v * g0.z; acc[3] += v * g0.w;
    acc[4] += v * g1.x; acc[5] += v * g1.y; acc[6] += v * g1.z; acc[7] += v * g1.w;
  }
#pragma unroll
  for (int e = 0; e < E_NUM; ++e)
    for (int off = 16; off > 0; off >>= 1)
      acc[e] += __shfl_xor(acc[e], off, 32);

  if (lane == 0) {
    float l[E_NUM], p[E_NUM];
    float m = -1e30f;
#pragma unroll
    for (int e = 0; e < E_NUM; ++e) { l[e] = acc[e] + gb[e]; m = fmaxf(m, l[e]); }
    float s = 0.0f;
#pragma unroll
    for (int e = 0; e < E_NUM; ++e) { p[e] = expf(l[e] - m); s += p[e]; }
    float inv = 1.0f / s;
    float ent = 0.0f;
#pragma unroll
    for (int e = 0; e < E_NUM; ++e) { p[e] *= inv; ent -= p[e] * logf(p[e] + 1e-10f); }

    int i1 = 0; float v1 = p[0];
#pragma unroll
    for (int e = 1; e < E_NUM; ++e) if (p[e] > v1) { v1 = p[e]; i1 = e; }
    int i2 = -1; float v2 = -1.0f;
#pragma unroll
    for (int e = 0; e < E_NUM; ++e) if (e != i1 && p[e] > v2) { v2 = p[e]; i2 = e; }

    atomicAdd(ent_accum, ent);
    int pos1 = atomicAdd(&cnt[i1], 1);
    tok_list[i1 * N + pos1] = tok; prob_list[i1 * N + pos1] = v1;
    int pos2 = atomicAdd(&cnt[i2], 1);
    tok_list[i2 * N + pos2] = tok; prob_list[i2 * N + pos2] = v2;
  }
}

// ---- grouped expert GEMM: WMMA f16/f32, double-buffered async global->LDS pipeline
__global__ __launch_bounds__(256) void moe_gemm(
    const unsigned short* __restrict__ xh,   // [N][D] f16
    const unsigned short* __restrict__ wt,   // [E][H][D] f16
    const float* __restrict__ eb,            // [E][H]
    const int* __restrict__ tok_list,        // [E][N]
    const float* __restrict__ prob_list,     // [E][N]
    const int* __restrict__ cnt,             // [E]
    float* __restrict__ out, int N) {
  int e = blockIdx.z;
  int ce = cnt[e];
  int row0 = blockIdx.y * BM;
  if (row0 >= ce) return;
  int col0 = blockIdx.x * BN;

  __shared__ alignas(16) unsigned short sA[2][BM * LDK2];
  __shared__ alignas(16) unsigned short sB[2][BN * LDK2];
  __shared__ int   sTok[BM];
  __shared__ float sPr[BM];

  int tid = threadIdx.x;
  if (tid < BM) {
    int r = row0 + tid;
    if (r < ce) { sTok[tid] = tok_list[e * N + r]; sPr[tid] = prob_list[e * N + r]; }
    else        { sTok[tid] = 0;                   sPr[tid] = 0.0f; }
  }
  __syncthreads();

  // ---- per-thread async copy chunk plan: 4 A-chunks + 4 B-chunks of 8 halves/stage
  const unsigned short* aSrc[4]; unsigned aOff[4];
  const unsigned short* bSrc[4]; unsigned bOff[4];
#pragma unroll
  for (int q = 0; q < 4; ++q) {
    int c = tid + 256 * q;          // 0..1023
    int m = c >> 3;                 // row in tile (0..127)
    int kc = (c & 7) * 8;           // half-offset within stage (0..56)
    aSrc[q] = xh + (size_t)sTok[m] * D_DIM + kc;
    aOff[q] = (unsigned)(m * LDK2 + kc) * 2u;      // byte offset inside buffer
    bSrc[q] = wt + ((size_t)e * H_DIM + col0 + m) * D_DIM + kc;
    bOff[q] = (unsigned)(m * LDK2 + kc) * 2u;
  }
  unsigned aBase[2] = { lds_off(&sA[0][0]), lds_off(&sA[1][0]) };
  unsigned bBase[2] = { lds_off(&sB[0][0]), lds_off(&sB[1][0]) };

  auto issue_stage = [&](int k0, int buf) {
#pragma unroll
    for (int q = 0; q < 4; ++q) async_copy_b128(aBase[buf] + aOff[q], aSrc[q] + k0);
#pragma unroll
    for (int q = 0; q < 4; ++q) async_copy_b128(bBase[buf] + bOff[q], bSrc[q] + k0);
  };

  int lane = tid & 31;
  int wv = tid >> 5;
  int wm = wv >> 1;            // 0..3  -> 32-row stripe
  int wn = wv & 1;             // 0..1  -> 64-col stripe
  int lr = lane & 15;
  int kb = (lane >> 4) * 8;    // K base: lanes 0-15 -> 0, lanes 16-31 -> 8

  v8f zero = {};
  v8f acc[2][4];
#pragma unroll
  for (int i = 0; i < 2; ++i)
#pragma unroll
    for (int j = 0; j < 4; ++j) acc[i][j] = zero;

  issue_stage(0, 0);                      // prologue: stage 0 in flight

  for (int s = 0; s < NSTAGE; ++s) {
    int cur = s & 1;
    if (s + 1 < NSTAGE) {
      issue_stage((s + 1) * BKS, (s + 1) & 1);     // prefetch next stage
      asm volatile("s_wait_asynccnt 0x8" ::: "memory");  // stage s (older 8) landed
    } else {
      asm volatile("s_wait_asynccnt 0x0" ::: "memory");
    }
    __syncthreads();                      // publish stage s tile to all waves

    const unsigned short* bufA = sA[cur];
    const unsigned short* bufB = sB[cur];
#pragma unroll
    for (int half = 0; half < 2; ++half) {
      int kk = half * 32;
      v16h afrag[2], bfrag[4];
#pragma unroll
      for (int i = 0; i < 2; ++i) {
        int m = wm * 32 + i * 16 + lr;
        union { uint4 u[2]; v16h h; } t;
        t.u[0] = *(const uint4*)(&bufA[m * LDK2 + kk + kb]);
        t.u[1] = *(const uint4*)(&bufA[m * LDK2 + kk + kb + 16]);
        afrag[i] = t.h;
      }
#pragma unroll
      for (int j = 0; j < 4; ++j) {
        int n = wn * 64 + j * 16 + lr;
        union { uint4 u[2]; v16h h; } t;
        t.u[0] = *(const uint4*)(&bufB[n * LDK2 + kk + kb]);
        t.u[1] = *(const uint4*)(&bufB[n * LDK2 + kk + kb + 16]);
        bfrag[j] = t.h;
      }
#pragma unroll
      for (int i = 0; i < 2; ++i)
#pragma unroll
        for (int j = 0; j < 4; ++j)
          acc[i][j] = __builtin_amdgcn_wmma_f32_16x16x32_f16(
              false, afrag[i], false, bfrag[j], (short)0, acc[i][j], false, false);
    }
    __syncthreads();                      // all waves done reading before overwrite
  }

  // epilogue: out[token, col] += prob * (acc + bias)
#pragma unroll
  for (int i = 0; i < 2; ++i) {
#pragma unroll
    for (int j = 0; j < 4; ++j) {
      int rbase = wm * 32 + i * 16 + (lane >> 4) * 8;
      int col = col0 + wn * 64 + j * 16 + lr;
      float bias = eb[e * H_DIM + col];
#pragma unroll
      for (int v = 0; v < 8; ++v) {
        int row = rbase + v;
        float val = sPr[row] * (acc[i][j][v] + bias);
        atomicAdd(out + (size_t)sTok[row] * H_DIM + col, val);
      }
    }
  }
}

// ---------------------------------------------------------------- aux loss scalar
__global__ void loss_kernel(const int* __restrict__ cnt, const float* __restrict__ ent,
                            float* __restrict__ out, int N) {
  if (threadIdx.x == 0 && blockIdx.x == 0) {
    float s = 0.1f * (*ent / (float)N);
    for (int e = 0; e < E_NUM; ++e) {
      float u = (float)cnt[e] / (float)N - 0.3f;
      if (u > 0.0f) s += u;
    }
    out[(size_t)N * H_DIM] = s;
  }
}

extern "C" void kernel_launch(void* const* d_in, const int* in_sizes, int n_in,
                              void* d_out, int out_size, void* d_ws, size_t ws_size,
                              hipStream_t stream) {
  const float* x  = (const float*)d_in[0];   // [B,S,D]
  const float* gw = (const float*)d_in[1];   // [D,E]
  const float* gb = (const float*)d_in[2];   // [E]
  const float* ew = (const float*)d_in[3];   // [E,D,H]
  const float* eb = (const float*)d_in[4];   // [E,H]
  float* out = (float*)d_out;
  int N = in_sizes[0] / D_DIM;               // 16384

  // workspace layout
  char* ws = (char*)d_ws;
  unsigned short* xh = (unsigned short*)ws;                               // N*D f16
  unsigned short* wt = (unsigned short*)(ws + (size_t)N * D_DIM * 2);     // E*H*D f16
  char* p = ws + (size_t)N * D_DIM * 2 + (size_t)E_NUM * H_DIM * D_DIM * 2;
  int*   tok_list  = (int*)p;   p += (size_t)E_NUM * N * 4;
  float* prob_list = (float*)p; p += (size_t)E_NUM * N * 4;
  int*   cnt       = (int*)p;   p += E_NUM * 4;
  float* ent       = (float*)p;

  size_t total = (size_t)N * H_DIM + 1;
  zero_kernel<<<(unsigned)((total + 255) / 256), 256, 0, stream>>>(out, total, cnt, ent);

  dim3 tg(H_DIM / 32, D_DIM / 32, E_NUM);
  wcvt_kernel<<<tg, 256, 0, stream>>>(ew, wt);

  gate_kernel<<<(N * 32 + 255) / 256, 256, 0, stream>>>(
      x, gw, gb, xh, tok_list, prob_list, cnt, ent, N);

  dim3 gg(H_DIM / BN, (N + BM - 1) / BM, E_NUM);
  moe_gemm<<<gg, 256, 0, stream>>>(xh, wt, eb, tok_list, prob_list, cnt, out, N);

  loss_kernel<<<1, 1, 0, stream>>>(cnt, ent, out, N);
}
